// Block_84198538871557
// MI455X (gfx1250) — matrix operand order repaired and checked by reading, not compiled, not run on hardware
//
#include <hip/hip_runtime.h>

// ---------------- problem constants (from reference) ----------------
constexpr int T     = 2048;
constexpr int C     = 2048;
constexpr int H     = 16;
constexpr int HS    = 128;
constexpr int AT    = 20;
constexpr int INTER = 5632;
constexpr float SCALE = 0.08838834764831845f;   // 1/sqrt(128)
constexpr float EPS   = 1e-5f;

// ---------------- WMMA types ----------------
typedef __attribute__((ext_vector_type(16))) __bf16          v16bf;
typedef __attribute__((ext_vector_type(16))) unsigned short  v16u;
typedef __attribute__((ext_vector_type(8)))  float           v8f;

union FragU { uint4 q[2]; v16u v; };

__device__ inline v8f wmma_bf16(v16u a, v16u b, v8f c) {
    return __builtin_amdgcn_wmma_f32_16x16x32_bf16(
        false, __builtin_bit_cast(v16bf, a),
        false, __builtin_bit_cast(v16bf, b),
        (short)0, c, false, false);
}

__device__ inline unsigned short f2bf(float f) {
    unsigned int u = __builtin_bit_cast(unsigned int, f);
    u += 0x7FFFu + ((u >> 16) & 1u);
    return (unsigned short)(u >> 16);
}

__device__ inline v8f vzero8() { v8f z = {0.f,0.f,0.f,0.f,0.f,0.f,0.f,0.f}; return z; }

// CDNA5 async global->LDS copy (ASYNCcnt tracked). 16B per lane.
// offset: applies to BOTH the global and the LDS address (ISA 15.18.3).
__device__ inline void async_copy_b128(unsigned lds_addr, const void* gptr) {
    asm volatile("global_load_async_to_lds_b128 %0, %1, off"
                 :: "v"(lds_addr), "v"(gptr) : "memory");
}
__device__ inline void async_copy_b128_off16(unsigned lds_addr, const void* gptr) {
    asm volatile("global_load_async_to_lds_b128 %0, %1, off offset:16"
                 :: "v"(lds_addr), "v"(gptr) : "memory");
}
__device__ inline void wait_async0() {
    asm volatile("s_wait_asynccnt 0x0" ::: "memory");
}
__device__ inline unsigned lds_off32(const void* p) {
    // generic LDS pointer: low 32 bits are the wave-relative LDS byte offset
    return (unsigned)(unsigned long long)p;
}

// =====================================================================
// fp32 -> bf16 cast (weights / activations)
// =====================================================================
__global__ __launch_bounds__(256) void castk(const float* __restrict__ in,
                                             unsigned short* __restrict__ out,
                                             size_t n) {
    size_t i = (size_t)blockIdx.x * 256 + threadIdx.x;
    size_t stride = (size_t)gridDim.x * 256;
    for (; i < n; i += stride) out[i] = f2bf(in[i]);
}

// sink k/v: [H,AT,HS] f32 -> [H,32,HS] bf16 zero-padded
__global__ __launch_bounds__(256) void cast_sink_k(const float* __restrict__ in,
                                                   unsigned short* __restrict__ out) {
    int i = blockIdx.x * 256 + threadIdx.x;            // H*32*HS = 65536
    int c = i & (HS - 1);
    int r = (i >> 7) & 31;
    int h = i >> 12;
    out[i] = (r < AT) ? f2bf(in[((size_t)h * AT + r) * HS + c]) : (unsigned short)0;
}

// =====================================================================
// RMSNorm: one block per row of C=2048, writes bf16
// =====================================================================
__global__ __launch_bounds__(256) void rmsnorm_k(const float* __restrict__ x,
                                                 const float* __restrict__ w,
                                                 unsigned short* __restrict__ out) {
    int t = blockIdx.x;
    int tid = threadIdx.x;
    const float* row = x + (size_t)t * C;
    float s = 0.f;
    for (int i = tid; i < C; i += 256) { float v = row[i]; s += v * v; }
    __shared__ float red[256];
    red[tid] = s;
    __syncthreads();
    for (int off = 128; off > 0; off >>= 1) {
        if (tid < off) red[tid] += red[tid + off];
        __syncthreads();
    }
    float r = rsqrtf(red[0] / (float)C + EPS);
    for (int i = tid; i < C; i += 256)
        out[(size_t)t * C + i] = f2bf(row[i] * r * w[i]);
}

// =====================================================================
// RoPE + QKV split: qkv f32 [T, 3C] -> q,k (roped), v bf16 [H][T][HS]
// =====================================================================
__global__ __launch_bounds__(128) void rope_k(const float* __restrict__ qkv,
                                              const float* __restrict__ cosb,
                                              const float* __restrict__ sinb,
                                              unsigned short* __restrict__ qb,
                                              unsigned short* __restrict__ kb,
                                              unsigned short* __restrict__ vb) {
    int t = blockIdx.x, h = blockIdx.y, hs = threadIdx.x;
    const float* base = qkv + (size_t)t * (3 * C) + (size_t)h * (3 * HS);
    float qv = base[hs], kv = base[HS + hs], vv = base[2 * HS + hs];
    float c = cosb[(size_t)t * HS + hs], s = sinb[(size_t)t * HS + hs];
    float qr = (hs < 64) ? -base[hs + 64]      : base[hs - 64];
    float kr = (hs < 64) ? -base[HS + hs + 64] : base[HS + hs - 64];
    size_t o = ((size_t)h * T + t) * HS + hs;
    qb[o] = f2bf(qv * c + qr * s);
    kb[o] = f2bf(kv * c + kr * s);
    vb[o] = f2bf(vv);
}

// =====================================================================
// bf16 WMMA GEMM: out[M,N] = A[M,K] @ B[N,K]^T  (both K-major, bf16)
// MODE 0: store f32     MODE 2: store f32 acc + resid
// block 256 (8 waves), tile 128x128, BK=32
// Double-buffered LDS fed by global_load_async_to_lds_b128 (ASYNCcnt).
// =====================================================================
template <int MODE>
__global__ __launch_bounds__(256) void gemm_bf16_k(const unsigned short* __restrict__ A,
                                                   const unsigned short* __restrict__ Bw,
                                                   float* __restrict__ outF,
                                                   const float* __restrict__ resid,
                                                   int M, int N, int K) {
    __shared__ unsigned short As[2][128][40];
    __shared__ unsigned short Bs[2][128][40];
    const int tid  = threadIdx.x;
    const int wave = tid >> 5, lane = tid & 31, l = lane & 15, hi = lane >> 4;
    const int m0 = blockIdx.y * 128, n0 = blockIdx.x * 128;
    const int wm = (wave >> 2) * 64, wn = (wave & 3) * 32;

    v8f acc[4][2];
#pragma unroll
    for (int i = 0; i < 4; i++)
#pragma unroll
        for (int j = 0; j < 2; j++) acc[i][j] = vzero8();

    const int r  = tid >> 1;
    const int cH = (tid & 1) * 16;

    // async prefetch of one 128x32 A-tile + B-tile into LDS buffer `buf`
    auto issue_tile = [&](int k0, int buf) {
        const unsigned short* ap = A  + (size_t)(m0 + r) * K + k0 + cH;
        const unsigned short* bp = Bw + (size_t)(n0 + r) * K + k0 + cH;
        unsigned la = lds_off32(&As[buf][r][cH]);
        unsigned lb = lds_off32(&Bs[buf][r][cH]);
        async_copy_b128(la, ap);
        async_copy_b128_off16(la, ap);     // offset:16 hits both LDS+global
        async_copy_b128(lb, bp);
        async_copy_b128_off16(lb, bp);
    };

    const int nk = K >> 5;
    issue_tile(0, 0);
    wait_async0();
    __syncthreads();

    for (int kt = 0; kt < nk; kt++) {
        const int buf = kt & 1;
        if (kt + 1 < nk) issue_tile((kt + 1) << 5, buf ^ 1);

        v16u af[4], bfg[2];
#pragma unroll
        for (int i = 0; i < 4; i++) {
            int row = wm + i * 16 + l;
            FragU f;
            f.q[0] = *reinterpret_cast<const uint4*>(&As[buf][row][hi * 8]);
            f.q[1] = *reinterpret_cast<const uint4*>(&As[buf][row][hi * 8 + 16]);
            af[i] = f.v;
        }
#pragma unroll
        for (int j = 0; j < 2; j++) {
            int col = wn + j * 16 + l;
            FragU f;
            f.q[0] = *reinterpret_cast<const uint4*>(&Bs[buf][col][hi * 8]);
            f.q[1] = *reinterpret_cast<const uint4*>(&Bs[buf][col][hi * 8 + 16]);
            bfg[j] = f.v;
        }
#pragma unroll
        for (int i = 0; i < 4; i++)
#pragma unroll
            for (int j = 0; j < 2; j++)
                acc[i][j] = wmma_bf16(af[i], bfg[j], acc[i][j]);

        wait_async0();        // this wave's prefetch landed
        __syncthreads();      // everyone done reading buf, prefetch visible
    }

#pragma unroll
    for (int i = 0; i < 4; i++)
#pragma unroll
        for (int j = 0; j < 2; j++)
#pragma unroll
            for (int e = 0; e < 8; e++) {
                int row = m0 + wm + i * 16 + e + hi * 8;
                int col = n0 + wn + j * 16 + l;
                size_t idx = (size_t)row * N + col;
                float v = acc[i][j][e];
                if constexpr (MODE == 2) v += resid[idx];
                outF[idx] = v;
            }
}

// =====================================================================
// Flash attention with WMMA (causal + sink pass, fused gating)
// block = 256 threads (8 waves), grid = (T/128, H); wave owns 16 q-rows
// =====================================================================
__device__ inline void attn_tile(const v16u qa[4],
                                 unsigned short (*Ks)[136],
                                 unsigned short (*Vs)[40],
                                 unsigned short (*Psw)[40],
                                 v8f O[8], float m[8], float lacc[8],
                                 int wq, int k0, int l, int hi,
                                 bool causal, int nvalid) {
    v8f s0 = vzero8(), s1 = vzero8();
#pragma unroll
    for (int c = 0; c < 4; c++) {
        FragU f;
        f.q[0] = *reinterpret_cast<const uint4*>(&Ks[l][c * 32 + hi * 8]);
        f.q[1] = *reinterpret_cast<const uint4*>(&Ks[l][c * 32 + hi * 8 + 16]);
        v16u b0 = f.v;
        f.q[0] = *reinterpret_cast<const uint4*>(&Ks[16 + l][c * 32 + hi * 8]);
        f.q[1] = *reinterpret_cast<const uint4*>(&Ks[16 + l][c * 32 + hi * 8 + 16]);
        v16u b1 = f.v;
        s0 = wmma_bf16(qa[c], b0, s0);
        s1 = wmma_bf16(qa[c], b1, s1);
    }
#pragma unroll
    for (int e = 0; e < 8; e++) {
        int qrow = wq + e + hi * 8;
        int key0 = k0 + l, key1 = k0 + 16 + l;
        float g0 = s0[e] * SCALE, g1 = s1[e] * SCALE;
        bool v0 = causal ? (key0 <= qrow) : (l < nvalid);
        bool v1 = causal ? (key1 <= qrow) : (16 + l < nvalid);
        if (!v0) g0 = -1e30f;
        if (!v1) g1 = -1e30f;
        float mt = fmaxf(g0, g1);
        for (int off = 1; off < 16; off <<= 1) mt = fmaxf(mt, __shfl_xor(mt, off));
        float mnew  = fmaxf(m[e], mt);
        float alpha = __expf(m[e] - mnew);
        float p0 = __expf(g0 - mnew), p1 = __expf(g1 - mnew);
        float rs = p0 + p1;
        for (int off = 1; off < 16; off <<= 1) rs += __shfl_xor(rs, off);
        lacc[e] = lacc[e] * alpha + rs;
        m[e] = mnew;
#pragma unroll
        for (int j = 0; j < 8; j++) O[j][e] *= alpha;
        Psw[e + hi * 8][l]      = f2bf(p0);
        Psw[e + hi * 8][16 + l] = f2bf(p1);
    }
    // P (C-layout) -> A-layout via per-wave LDS staging (DS in-order per wave)
    FragU f;
    f.q[0] = *reinterpret_cast<const uint4*>(&Psw[l][hi * 8]);
    f.q[1] = *reinterpret_cast<const uint4*>(&Psw[l][hi * 8 + 16]);
    v16u pa = f.v;
#pragma unroll
    for (int j = 0; j < 8; j++) {
        FragU g;
        g.q[0] = *reinterpret_cast<const uint4*>(&Vs[j * 16 + l][hi * 8]);
        g.q[1] = *reinterpret_cast<const uint4*>(&Vs[j * 16 + l][hi * 8 + 16]);
        O[j] = wmma_bf16(pa, g.v, O[j]);
    }
}

__global__ __launch_bounds__(256) void attn_k(const unsigned short* __restrict__ Qg,
                                              const unsigned short* __restrict__ Kg,
                                              const unsigned short* __restrict__ Vg,
                                              const unsigned short* __restrict__ AKg,
                                              const unsigned short* __restrict__ AVg,
                                              const float* __restrict__ gating,
                                              unsigned short* __restrict__ Y) {
    const int h = blockIdx.y, qtile = blockIdx.x;
    const int tid = threadIdx.x, wave = tid >> 5, lane = tid & 31;
    const int l = lane & 15, hi = lane >> 4;
    const int qbase = qtile * 128;
    const int wq = qbase + wave * 16;

    __shared__ unsigned short Ks[32][136];   // [key][hs]
    __shared__ unsigned short Vs[128][40];   // transposed [hs][key]
    __shared__ unsigned short Ps[8][16][40]; // per-wave P staging

    // Q fragments (A-layout), 4 chunks of K=32 across HS=128
    v16u qa[4];
    {
        const unsigned short* qp = Qg + ((size_t)h * T + wq + l) * HS;
#pragma unroll
        for (int c = 0; c < 4; c++) {
            FragU f;
            f.q[0] = *reinterpret_cast<const uint4*>(qp + c * 32 + hi * 8);
            f.q[1] = *reinterpret_cast<const uint4*>(qp + c * 32 + hi * 8 + 16);
            qa[c] = f.v;
        }
    }

    v8f O1[8], O2[8];
    float m1[8], l1[8], m2[8], l2[8];
#pragma unroll
    for (int j = 0; j < 8; j++) { O1[j] = vzero8(); O2[j] = vzero8(); }
#pragma unroll
    for (int e = 0; e < 8; e++) { m1[e] = -1e30f; l1[e] = 0.f; m2[e] = -1e30f; l2[e] = 0.f; }

    const int rldr = tid >> 3, cldr = (tid & 7) * 16;   // K tile loader
    const int vkey = tid & 31, vhsg = tid >> 5;          // V transpose loader
    const unsigned ksaddr = lds_off32(&Ks[rldr][cldr]);

    // ---------------- causal pass ----------------
    for (int k0 = 0; k0 < qbase + 128; k0 += 32) {
        {
            // K tile: async global->LDS (ASYNCcnt), 32B per thread
            const unsigned short* kp = Kg + ((size_t)h * T + k0 + rldr) * HS + cldr;
            async_copy_b128(ksaddr, kp);
            async_copy_b128_off16(ksaddr, kp);
            // V tile: load + transpose into Vs[hs][key]
            const unsigned short* vp = Vg + ((size_t)h * T + k0 + vkey) * HS + vhsg * 16;
#pragma unroll
            for (int j = 0; j < 16; j++) Vs[vhsg * 16 + j][vkey] = vp[j];
            wait_async0();
        }
        __syncthreads();
        attn_tile(qa, Ks, Vs, Ps[wave], O1, m1, l1, wq, k0, l, hi, true, 0);
        __syncthreads();
    }

    // ---------------- sink pass (AT=20 keys, padded to 32) ----------------
    {
        const unsigned short* kp = AKg + ((size_t)h * 32 + rldr) * HS + cldr;
        async_copy_b128(ksaddr, kp);
        async_copy_b128_off16(ksaddr, kp);
        const unsigned short* vp = AVg + ((size_t)h * 32 + vkey) * HS + vhsg * 16;
#pragma unroll
        for (int j = 0; j < 16; j++) Vs[vhsg * 16 + j][vkey] = vp[j];
        wait_async0();
    }
    __syncthreads();
    attn_tile(qa, Ks, Vs, Ps[wave], O2, m2, l2, wq, 0, l, hi, false, AT);

    // ---------------- epilogue: y = O1/l1 + g * O2/l2, bf16 [T, C] ----------------
    const float g = gating[h];
#pragma unroll
    for (int j = 0; j < 8; j++)
#pragma unroll
        for (int e = 0; e < 8; e++) {
            int t   = wq + e + hi * 8;
            int col = h * HS + j * 16 + l;
            float y = O1[j][e] / l1[e] + g * (O2[j][e] / l2[e]);
            Y[(size_t)t * C + col] = f2bf(y);
        }
}

// =====================================================================
// SiLU gate: out_bf16 = silu(a) * b
// =====================================================================
__global__ __launch_bounds__(256) void silu_k(const float* __restrict__ a,
                                              const float* __restrict__ b,
                                              unsigned short* __restrict__ out,
                                              size_t n) {
    size_t i = (size_t)blockIdx.x * 256 + threadIdx.x;
    size_t stride = (size_t)gridDim.x * 256;
    for (; i < n; i += stride) {
        float x = a[i];
        float s = x / (1.f + __expf(-x));
        out[i] = f2bf(s * b[i]);
    }
}

// =====================================================================
// host-side orchestration
// =====================================================================
extern "C" void kernel_launch(void* const* d_in, const int* in_sizes, int n_in,
                              void* d_out, int out_size, void* d_ws, size_t ws_size,
                              hipStream_t stream) {
    (void)in_sizes; (void)n_in; (void)out_size; (void)ws_size;
    const float* x      = (const float*)d_in[0];
    const float* cosb   = (const float*)d_in[1];
    const float* sinb   = (const float*)d_in[2];
    // d_in[3] = mask (implicit causal, unused)
    const float* ak     = (const float*)d_in[4];
    const float* av     = (const float*)d_in[5];
    const float* w_qkv  = (const float*)d_in[6];
    const float* w_proj = (const float*)d_in[7];
    const float* gating = (const float*)d_in[8];
    const float* n1w    = (const float*)d_in[9];
    const float* n2w    = (const float*)d_in[10];
    const float* w_fc1  = (const float*)d_in[11];
    const float* w_fc2  = (const float*)d_in[12];
    const float* w_mlp  = (const float*)d_in[13];
    float* out = (float*)d_out;

    // workspace carve-up (~312 MB total)
    char* ws = (char*)d_ws;
    size_t off = 0;
    auto alloc = [&](size_t bytes) -> char* {
        char* p = ws + off;
        off += (bytes + 255) & ~(size_t)255;
        return p;
    };
    unsigned short* nbf    = (unsigned short*)alloc((size_t)T * C * 2);
    unsigned short* wqkvb  = (unsigned short*)alloc((size_t)3 * C * C * 2);
    unsigned short* wprojb = (unsigned short*)alloc((size_t)C * C * 2);
    unsigned short* wfc1b  = (unsigned short*)alloc((size_t)INTER * C * 2);
    unsigned short* wfc2b  = (unsigned short*)alloc((size_t)INTER * C * 2);
    unsigned short* wmlpb  = (unsigned short*)alloc((size_t)INTER * C * 2);
    float*          qkvf   = (float*)alloc((size_t)T * 3 * C * 4);
    unsigned short* qb     = (unsigned short*)alloc((size_t)T * C * 2);
    unsigned short* kb     = (unsigned short*)alloc((size_t)T * C * 2);
    unsigned short* vb     = (unsigned short*)alloc((size_t)T * C * 2);
    unsigned short* akb    = (unsigned short*)alloc((size_t)H * 32 * HS * 2);
    unsigned short* avb    = (unsigned short*)alloc((size_t)H * 32 * HS * 2);
    unsigned short* yb     = (unsigned short*)alloc((size_t)T * C * 2);
    float*          xres   = (float*)alloc((size_t)T * C * 4);
    float*          f1     = (float*)alloc((size_t)T * INTER * 4);
    float*          f2g    = (float*)alloc((size_t)T * INTER * 4);
    unsigned short* h1b    = (unsigned short*)alloc((size_t)T * INTER * 2);

    // weight casts to bf16
    castk<<<4096, 256, 0, stream>>>(w_qkv,  wqkvb, (size_t)3 * C * C);
    castk<<<4096, 256, 0, stream>>>(w_proj, wprojb, (size_t)C * C);
    castk<<<4096, 256, 0, stream>>>(w_fc1,  wfc1b, (size_t)INTER * C);
    castk<<<4096, 256, 0, stream>>>(w_fc2,  wfc2b, (size_t)INTER * C);
    castk<<<4096, 256, 0, stream>>>(w_mlp,  wmlpb, (size_t)INTER * C);
    cast_sink_k<<<(H * 32 * HS) / 256, 256, 0, stream>>>(ak, akb);
    cast_sink_k<<<(H * 32 * HS) / 256, 256, 0, stream>>>(av, avb);

    // n1 = rmsnorm(x)
    rmsnorm_k<<<T, 256, 0, stream>>>(x, n1w, nbf);

    // qkv = n1 @ w_qkv^T   [2048 x 6144]
    gemm_bf16_k<0><<<dim3(3 * C / 128, T / 128), 256, 0, stream>>>(
        nbf, wqkvb, qkvf, nullptr, T, 3 * C, C);

    // rope + split
    rope_k<<<dim3(T, H), 128, 0, stream>>>(qkvf, cosb, sinb, qb, kb, vb);

    // flash attention (causal + sinks + gating)
    attn_k<<<dim3(T / 128, H), 256, 0, stream>>>(qb, kb, vb, akb, avb, gating, yb);

    // x_res = x + y @ w_proj^T
    gemm_bf16_k<2><<<dim3(C / 128, T / 128), 256, 0, stream>>>(
        yb, wprojb, xres, x, T, C, C);

    // n2 = rmsnorm(x_res)
    rmsnorm_k<<<T, 256, 0, stream>>>(xres, n2w, nbf);

    // mlp up projections
    gemm_bf16_k<0><<<dim3(INTER / 128, T / 128), 256, 0, stream>>>(
        nbf, wfc1b, f1, nullptr, T, INTER, C);
    gemm_bf16_k<0><<<dim3(INTER / 128, T / 128), 256, 0, stream>>>(
        nbf, wfc2b, f2g, nullptr, T, INTER, C);

    // h1 = silu(f1) * f2
    silu_k<<<4096, 256, 0, stream>>>(f1, f2g, h1b, (size_t)T * INTER);

    // out = x_res + h1 @ w_mlp_proj^T
    gemm_bf16_k<2><<<dim3(C / 128, T / 128), 256, 0, stream>>>(
        h1b, wmlpb, out, xres, T, C, INTER);
}
